// MultiHeadAttention_28449863369041
// MI455X (gfx1250) — compile-verified, matrix operand order checked by reading, not run on hardware
//
#include <hip/hip_runtime.h>
#include <hip/hip_bf16.h>

// ---------------------------------------------------------------------------
// MultiHeadAttention forward for MI455X (gfx1250, wave32, WMMA + TDM).
// B=2, S=2048, E=2048, H=16, D=128.  All GEMM-class math in bf16 WMMA
// (v_wmma_f32_16x16x32_bf16), fp32 accumulation, flash-attention style
// online softmax.  GEMM tiles staged into LDS by the Tensor Data Mover
// (tensor_load_to_lds + s_wait_tensorcnt) when the builtin is available.
// ---------------------------------------------------------------------------

#define MB_ 2
#define MS_ 2048
#define ME_ 2048
#define MH_ 16
#define MD_ 128

typedef __attribute__((ext_vector_type(16))) __bf16        v16bf;
typedef __attribute__((ext_vector_type(8)))  float         v8f;
typedef __attribute__((ext_vector_type(4)))  unsigned int  v4u;
typedef __attribute__((ext_vector_type(8)))  int           v8i;
typedef __attribute__((ext_vector_type(4)))  int           v4i;

#if defined(__has_builtin)
#if __has_builtin(__builtin_amdgcn_tensor_load_to_lds)
#define USE_TDM 1
#endif
#endif
#ifndef USE_TDM
#define USE_TDM 0
#endif

__device__ __forceinline__ unsigned short f2bf(float f) {
    unsigned int u = __float_as_uint(f);
    unsigned int r = u + 0x7FFFu + ((u >> 16) & 1u);   // round to nearest even
    return (unsigned short)(r >> 16);
}

__device__ __forceinline__ float half_reduce_max(float v) {
    v = fmaxf(v, __shfl_xor(v, 1));
    v = fmaxf(v, __shfl_xor(v, 2));
    v = fmaxf(v, __shfl_xor(v, 4));
    v = fmaxf(v, __shfl_xor(v, 8));
    return v;   // masks < 16 keep lanes 0-15 / 16-31 separate (wave32)
}
__device__ __forceinline__ float half_reduce_sum(float v) {
    v += __shfl_xor(v, 1);
    v += __shfl_xor(v, 2);
    v += __shfl_xor(v, 4);
    v += __shfl_xor(v, 8);
    return v;
}

#if USE_TDM
// Issue one TDM load of a 2-D bf16 tile (tile_x cols x tile_y rows, row
// stride `stride_elems`) from global memory into LDS at byte offset lds_off.
// D# bitfields per CDNA5 ISA ch.8 (group0: count/lds/global/type,
// group1: data_size, tensor_dim0/1, tile_dim0/1, tensor_dim0_stride).
// This toolchain exposes the 6-arg builtin:
//   (uint32x4 g0, int32x8 g1, int32x4 g2, int32x4 g3, int32x8 g4, i32 cpol)
__device__ __forceinline__ void tdm_load_2d(unsigned lds_off, const void* gptr,
                                            unsigned tile_x, unsigned tile_y,
                                            unsigned stride_elems) {
    unsigned long long ga = (unsigned long long)gptr;
    v4u g0;
    g0.x = 1u;                                        // count=1, user mode
    g0.y = lds_off;                                   // lds_addr (bytes)
    g0.z = (unsigned)(ga & 0xFFFFFFFFu);              // global_addr[31:0]
    g0.w = (unsigned)((ga >> 32) & 0x01FFFFFFu)       // global_addr[56:32]
         | (2u << 30);                                // type = 2 ("image")
    v8i g1;
    const unsigned td0 = stride_elems;                // tensor_dim0 (elems)
    const unsigned td1 = tile_y;                      // tensor_dim1 (rows)
    g1[0] = (int)0x00010000u;                         // data_size=1 -> 2 bytes
    g1[1] = (int)((td0 & 0xFFFFu) << 16);             // tensor_dim0[15:0]
    g1[2] = (int)((td0 >> 16) | ((td1 & 0xFFFFu) << 16));
    g1[3] = (int)((td1 >> 16) | (tile_x << 16));      // tile_dim0
    g1[4] = (int)(tile_y & 0xFFFFu);                  // tile_dim1 (tile_dim2=0)
    g1[5] = (int)stride_elems;                        // tensor_dim0_stride lo32
    g1[6] = 0;                                        // stride hi16 | dim1_stride lo
    g1[7] = 0;
    v4i z4 = {};
    v8i z8 = {};
    __builtin_amdgcn_tensor_load_to_lds(g0, g1, z4, z4, z8, 0);
}
#endif

// ---------------------------------------------------------------------------
// fp32 -> bf16 cast
// ---------------------------------------------------------------------------
__global__ void cast_bf16_kernel(const float* __restrict__ src,
                                 unsigned short* __restrict__ dst, int n) {
    int i = blockIdx.x * blockDim.x + threadIdx.x;
    int stride = gridDim.x * blockDim.x;
    for (; i < n; i += stride) dst[i] = f2bf(src[i]);
}

// ---------------------------------------------------------------------------
// Y[m][n] = sum_k A[m][k] * W[n][k]   (A: MxK bf16, W: NxK bf16)
// Block: 256 threads = 8 waves, 256x128 tile; wave computes 32x128.
// K staged 64-wide in LDS (TDM when available) -> 32 WMMAs/wave/K-step.
// ---------------------------------------------------------------------------
template <bool F32OUT>
__global__ __launch_bounds__(256) void gemm_bf16_nt(
    const unsigned short* __restrict__ A,
    const unsigned short* __restrict__ W,
    void* __restrict__ outp, int M, int N, int K) {
    __shared__ unsigned short As[256 * 64];   // 32 KB, row stride 128 B
    __shared__ unsigned short Bs[128 * 64];   // 16 KB

    const int tid  = threadIdx.x;
    const int wave = tid >> 5;
    const int lane = tid & 31;
    const int col  = lane & 15;
    const int half = lane >> 4;
    const int m0 = blockIdx.y * 256;
    const int n0 = blockIdx.x * 128;

    v8f acc0[8] = {};
    v8f acc1[8] = {};

#pragma unroll 1
    for (int kb = 0; kb < K; kb += 64) {
#if USE_TDM
        if (wave == 0) {
            tdm_load_2d((unsigned)(unsigned long long)As,
                        &A[(size_t)m0 * K + kb], 64u, 256u, (unsigned)K);
            tdm_load_2d((unsigned)(unsigned long long)Bs,
                        &W[(size_t)n0 * K + kb], 64u, 128u, (unsigned)K);
            __builtin_amdgcn_s_wait_tensorcnt(0);
        }
#else
        {   // A: thread t copies its 64-element row (8 x 16B)
            const unsigned short* ga = &A[(size_t)(m0 + tid) * K + kb];
#pragma unroll
            for (int j = 0; j < 8; ++j)
                *(uint4*)&As[tid * 64 + j * 8] = *(const uint4*)&ga[j * 8];
            // B: thread pair per row, 32 elements each (4 x 16B)
            const unsigned short* gb =
                &W[(size_t)(n0 + (tid >> 1)) * K + kb + (tid & 1) * 32];
#pragma unroll
            for (int j = 0; j < 4; ++j)
                *(uint4*)&Bs[(tid >> 1) * 64 + (tid & 1) * 32 + j * 8] =
                    *(const uint4*)&gb[j * 8];
        }
#endif
        __syncthreads();

#pragma unroll
        for (int ks = 0; ks < 64; ks += 32) {
            v16bf a0 = *(const v16bf*)&As[(wave * 32 + col) * 64 + ks + half * 16];
            v16bf a1 = *(const v16bf*)&As[(wave * 32 + 16 + col) * 64 + ks + half * 16];
#pragma unroll
            for (int c = 0; c < 8; ++c) {
                v16bf bf = *(const v16bf*)&Bs[(c * 16 + col) * 64 + ks + half * 16];
                acc0[c] = __builtin_amdgcn_wmma_f32_16x16x32_bf16(
                    false, a0, false, bf, (short)0, acc0[c], false, false);
                acc1[c] = __builtin_amdgcn_wmma_f32_16x16x32_bf16(
                    false, a1, false, bf, (short)0, acc1[c], false, false);
            }
        }
        __syncthreads();
    }

    // C layout: VGPR r, lane l -> row = r + 8*(l>=16), col = l&15
#pragma unroll
    for (int c = 0; c < 8; ++c) {
#pragma unroll
        for (int r = 0; r < 8; ++r) {
            const int n = n0 + c * 16 + col;
            const int ma = m0 + wave * 32 + r + 8 * half;
            const int mb = ma + 16;
            if (F32OUT) {
                ((float*)outp)[(size_t)ma * N + n] = acc0[c][r];
                ((float*)outp)[(size_t)mb * N + n] = acc1[c][r];
            } else {
                ((unsigned short*)outp)[(size_t)ma * N + n] = f2bf(acc0[c][r]);
                ((unsigned short*)outp)[(size_t)mb * N + n] = f2bf(acc1[c][r]);
            }
        }
    }
}

// ---------------------------------------------------------------------------
// Flash attention: one block = (b, h, 64 q rows); 4 waves x 16 q rows.
// Key/value tiles of 32 processed per step; causal; online softmax.
// Q,K,V,O stored (b, s, h*D+d) bf16.
// ---------------------------------------------------------------------------
__global__ __launch_bounds__(128) void attn_kernel(
    const unsigned short* __restrict__ Q,
    const unsigned short* __restrict__ K,
    const unsigned short* __restrict__ V,
    unsigned short* __restrict__ O) {
    __shared__ unsigned short Kt[32 * 128];    // [key][d]      8 KB
    __shared__ unsigned short Vt[128 * 48];    // [d][key] pad  12 KB (stride 96 B)
    __shared__ unsigned short Pt[4 * 16 * 32]; // per-wave P    4 KB (stride 64 B)

    const int tid  = threadIdx.x;
    const int wave = tid >> 5;
    const int lane = tid & 31;
    const int col  = lane & 15;
    const int half = lane >> 4;

    const int qt = blockIdx.x;          // 0..31
    const int h  = blockIdx.y;          // 0..15
    const int b  = blockIdx.z;          // 0..1
    const int q0 = qt * 64;
    const int qw = q0 + wave * 16;      // wave's first q row

    // Q fragments: 16 rows x 128 d, kept in registers (4 x 16x32 A-frags)
    v16bf qf[4];
    {
        const size_t qrow = ((size_t)b * MS_ + (qw + col)) * ME_ + h * MD_;
#pragma unroll
        for (int f = 0; f < 4; ++f)
            qf[f] = *(const v16bf*)&Q[qrow + f * 32 + half * 16];
    }

    v8f o[8] = {};
    float mrow[8], lrow[8];
#pragma unroll
    for (int r = 0; r < 8; ++r) { mrow[r] = -3.0e38f; lrow[r] = 0.0f; }

    const float scale = 0.088388347648318447f;   // 1/sqrt(128)
    const int nkt = (q0 + 64) / 32;              // causal: keys <= q0+63
    unsigned short* Pw = &Pt[wave * 512];

#pragma unroll 1
    for (int kt = 0; kt < nkt; ++kt) {
        const int kb = kt * 32;
        // ---- stage K tile (row major) and V tile (transposed) ----
        for (int idx = tid; idx < 512; idx += 128) {
            const int e  = idx * 8;
            const int kr = e >> 7;       // key row 0..31
            const int dc = e & 127;      // d col, multiple of 8
            const size_t g = ((size_t)b * MS_ + (kb + kr)) * ME_ + h * MD_ + dc;
            *(uint4*)&Kt[kr * 128 + dc] = *(const uint4*)&K[g];
            uint4 pk = *(const uint4*)&V[g];
            const unsigned short* pv = (const unsigned short*)&pk;
#pragma unroll
            for (int j = 0; j < 8; ++j) Vt[(dc + j) * 48 + kr] = pv[j];
        }
        __syncthreads();

        // ---- scores: two 16x16 WMMA tiles over 32 keys ----
        v8f s0 = {}, s1 = {};
#pragma unroll
        for (int f = 0; f < 4; ++f) {
            v16bf kf0 = *(const v16bf*)&Kt[(col)      * 128 + f * 32 + half * 16];
            v16bf kf1 = *(const v16bf*)&Kt[(16 + col) * 128 + f * 32 + half * 16];
            s0 = __builtin_amdgcn_wmma_f32_16x16x32_bf16(false, qf[f], false, kf0,
                                                         (short)0, s0, false, false);
            s1 = __builtin_amdgcn_wmma_f32_16x16x32_bf16(false, qf[f], false, kf1,
                                                         (short)0, s1, false, false);
        }

        // ---- online softmax over the 32 new columns ----
        float alpha[8];
#pragma unroll
        for (int r = 0; r < 8; ++r) {
            const int qr = qw + r + 8 * half;
            float a  = s0[r] * scale + ((kb + col)      > qr ? -1.0e9f : 0.0f);
            float c2 = s1[r] * scale + ((kb + 16 + col) > qr ? -1.0e9f : 0.0f);
            float mx = half_reduce_max(fmaxf(a, c2));
            float mnew = fmaxf(mrow[r], mx);
            float al = __expf(mrow[r] - mnew);
            float p0 = __expf(a - mnew);
            float p1 = __expf(c2 - mnew);
            float ps = half_reduce_sum(p0 + p1);
            lrow[r] = lrow[r] * al + ps;
            mrow[r] = mnew;
            alpha[r] = al;
            // scatter P (C layout -> LDS row major, stride 32 elems = 64 B)
            Pw[(r + 8 * half) * 32 + col]      = f2bf(p0);
            Pw[(r + 8 * half) * 32 + 16 + col] = f2bf(p1);
        }

        // rescale running output by alpha (per row)
#pragma unroll
        for (int c = 0; c < 8; ++c)
#pragma unroll
            for (int r = 0; r < 8; ++r) o[c][r] *= alpha[r];

        // ---- PV: P (16x32) x V (32x128) -> accumulate into o ----
        v16bf pf = *(const v16bf*)&Pw[col * 32 + half * 16];
#pragma unroll
        for (int c = 0; c < 8; ++c) {
            v16bf vf = *(const v16bf*)&Vt[(c * 16 + col) * 48 + half * 16];
            o[c] = __builtin_amdgcn_wmma_f32_16x16x32_bf16(
                false, pf, false, vf, (short)0, o[c], false, false);
        }
        __syncthreads();   // before K/V tile is overwritten
    }

    // ---- normalize and store bf16 output ----
    float inv[8];
#pragma unroll
    for (int r = 0; r < 8; ++r) inv[r] = 1.0f / lrow[r];
#pragma unroll
    for (int c = 0; c < 8; ++c) {
#pragma unroll
        for (int r = 0; r < 8; ++r) {
            const size_t g = ((size_t)b * MS_ + (qw + r + 8 * half)) * ME_ +
                             h * MD_ + c * 16 + col;
            O[g] = f2bf(o[c][r] * inv[r]);
        }
    }
}

// ---------------------------------------------------------------------------
// Host launcher
// ---------------------------------------------------------------------------
extern "C" void kernel_launch(void* const* d_in, const int* in_sizes, int n_in,
                              void* d_out, int out_size, void* d_ws, size_t ws_size,
                              hipStream_t stream) {
    (void)in_sizes; (void)n_in; (void)out_size; (void)ws_size;
    const float* x  = (const float*)d_in[0];
    const float* Wq = (const float*)d_in[1];
    const float* Wk = (const float*)d_in[2];
    const float* Wv = (const float*)d_in[3];
    const float* Wo = (const float*)d_in[4];

    const size_t nx = (size_t)MB_ * MS_ * ME_;   // 8,388,608
    const size_t nw = (size_t)ME_ * ME_;         // 4,194,304

    unsigned short* ws  = (unsigned short*)d_ws;
    unsigned short* XB  = ws;
    unsigned short* WQB = XB + nx;
    unsigned short* WKB = WQB + nw;
    unsigned short* WVB = WKB + nw;
    unsigned short* WOB = WVB + nw;
    unsigned short* QB  = WOB + nw;
    unsigned short* KB  = QB + nx;
    unsigned short* VB  = KB + nx;
    unsigned short* AOB = VB + nx;   // total 112 MB of workspace

    // 1) fp32 -> bf16 casts
    cast_bf16_kernel<<<8192, 256, 0, stream>>>(x,  XB,  (int)nx);
    cast_bf16_kernel<<<4096, 256, 0, stream>>>(Wq, WQB, (int)nw);
    cast_bf16_kernel<<<4096, 256, 0, stream>>>(Wk, WKB, (int)nw);
    cast_bf16_kernel<<<4096, 256, 0, stream>>>(Wv, WVB, (int)nw);
    cast_bf16_kernel<<<4096, 256, 0, stream>>>(Wo, WOB, (int)nw);

    // 2) Q/K/V projections: y = x @ W^T   (M=4096, N=2048, K=2048)
    const int M = MB_ * MS_, N = ME_, Kd = ME_;
    dim3 ggrid(N / 128, M / 256);
    gemm_bf16_nt<false><<<ggrid, 256, 0, stream>>>(XB, WQB, QB, M, N, Kd);
    gemm_bf16_nt<false><<<ggrid, 256, 0, stream>>>(XB, WKB, KB, M, N, Kd);
    gemm_bf16_nt<false><<<ggrid, 256, 0, stream>>>(XB, WVB, VB, M, N, Kd);

    // 3) causal flash attention
    dim3 agrid(MS_ / 64, MH_, MB_);
    attn_kernel<<<agrid, 128, 0, stream>>>(QB, KB, VB, AOB);

    // 4) output projection -> fp32 d_out
    gemm_bf16_nt<true><<<ggrid, 256, 0, stream>>>(AOB, WOB, (float*)d_out, M, N, Kd);
}
